// Similarity_83133386981519
// MI455X (gfx1250) — compile-verified, matrix operand order checked by reading, not run on hardware
//
#include <hip/hip_runtime.h>
#include <math.h>

typedef __attribute__((ext_vector_type(16))) _Float16 v16h;
typedef __attribute__((ext_vector_type(8)))  float    v8f;

#define Bdim 8
#define Ndim 1024
#define Mdim 1024
#define THREADS 256
#define WAVES 8
#define ROWS_PER_WG 16
#define ROWS_PER_WAVE 2   /* ROWS_PER_WG / WAVES */
#define DROW 20           /* padded floats per D row in LDS scratch */

union V16U { v16h h; int i[8]; };

__device__ __forceinline__ int pk2(float a, float b) {
  union { _Float16 h[2]; int i; } u;
  u.h[0] = (_Float16)a; u.h[1] = (_Float16)b;
  return u.i;
}

__global__ __launch_bounds__(THREADS)
void similarity_kernel(const float* __restrict__ points,
                       const float* __restrict__ ctxg,
                       const float* __restrict__ dist,
                       const unsigned char* __restrict__ inter,
                       const int* __restrict__ gmaxd,
                       const float* __restrict__ W1,
                       const float* __restrict__ b1,
                       const float* __restrict__ W2,
                       const float* __restrict__ b2,
                       const float* __restrict__ Wa,
                       const float* __restrict__ ba,
                       float* __restrict__ out)
{
  __shared__ float ctx_s[Mdim * 8];          /* 32 KB: context padded to 8 f32/row */
  __shared__ float dscr[WAVES * 32 * DROW];  /* 20 KB: per-wave WMMA D scratch */

  const int tid  = threadIdx.x;
  const int wg   = blockIdx.x;
  const int b    = wg / (Ndim / ROWS_PER_WG);
  const int nbase = (wg % (Ndim / ROWS_PER_WG)) * ROWS_PER_WG;
  const int wave = tid >> 5;
  const int lane = tid & 31;
  const int col  = lane & 15;
  const int hf   = lane >> 4;

  /* ---- stage context[b] (1024 x 5 f32) into LDS, padded stride 8, async DMA ---- */
  {
    const float* g0 = ctxg + (size_t)b * Mdim * 5;
    for (int r = tid; r < Mdim; r += THREADS) {
      unsigned lds_off = (unsigned)(unsigned long long)(uintptr_t)&ctx_s[r * 8];
      unsigned long long ga = (unsigned long long)(uintptr_t)(g0 + (size_t)r * 5);
      asm volatile("global_load_async_to_lds_b32 %0, %1, off offset:0"  :: "v"(lds_off), "v"(ga) : "memory");
      asm volatile("global_load_async_to_lds_b32 %0, %1, off offset:4"  :: "v"(lds_off), "v"(ga) : "memory");
      asm volatile("global_load_async_to_lds_b32 %0, %1, off offset:8"  :: "v"(lds_off), "v"(ga) : "memory");
      asm volatile("global_load_async_to_lds_b32 %0, %1, off offset:12" :: "v"(lds_off), "v"(ga) : "memory");
      asm volatile("global_load_async_to_lds_b32 %0, %1, off offset:16" :: "v"(lds_off), "v"(ga) : "memory");
    }
    asm volatile("s_wait_asynccnt 0" ::: "memory");
  }
  __syncthreads();

  /* ---- build B matrix (f16 32x16): columns 0..3 = W1, column 4 = Wa, rest 0 ---- */
  V16U Bm;
  #pragma unroll
  for (int k = 0; k < 8; ++k) Bm.i[k] = 0;
  if (lane < 16) {
    float w0 = 0.f, w1 = 0.f, w2c = 0.f, w3 = 0.f, w4 = 0.f;
    if (lane < 4) {
      w0 = W1[0*4 + lane]; w1 = W1[1*4 + lane]; w2c = W1[2*4 + lane];
      w3 = W1[3*4 + lane]; w4 = W1[4*4 + lane];
    } else if (lane == 4) {
      w0 = Wa[0]; w1 = Wa[1]; w2c = Wa[2]; w3 = Wa[3]; w4 = Wa[4];
    }
    Bm.i[0] = pk2(w0, w1);
    Bm.i[1] = pk2(w2c, w3);
    Bm.i[2] = pk2(w4, 0.0f);
  }

  /* ---- bake biases into C: cols 0..3 = b1, col 4 = ba ---- */
  const float cv = (col < 4) ? b1[col] : ((col == 4) ? ba[0] : 0.0f);
  v8f cbias;
  #pragma unroll
  for (int r = 0; r < 8; ++r) cbias[r] = cv;

  const float w2v0 = W2[0], w2v1 = W2[1], w2v2 = W2[2], w2v3 = W2[3], b2v = b2[0];
  /* fold "tighten" into the threshold so the hot loop is branch-free */
  const float thr = (50.0f < (float)gmaxd[0]) ? 50.0f : 3.4e38f;
  const float inv0 = 1.0f / 50.0f, inv2 = 1.0f / 32.0f, inv4 = 0.31830988618379067f;

  float* wbase = &dscr[wave * 32 * DROW];

  for (int rr = 0; rr < ROWS_PER_WAVE; ++rr) {
    const int n = nbase + wave * ROWS_PER_WAVE + rr;
    const float* prow = points + ((size_t)b * Ndim + n) * 5;
    const float p0 = prow[0], p1 = prow[1], p2 = prow[2], p3 = prow[3], p4 = prow[4];
    const float* drow = dist + ((size_t)b * Ndim + n) * Mdim;
    const unsigned char* irow = inter + ((size_t)b * Ndim + n) * Mdim;

    float denom = 0.f, numer = 0.f;

    for (int step = 0; step < Mdim / 32; ++step) {
      const int m = step * 32 + lane;

      /* issue the two independent global streams up front (unconditional) */
      const float dv = drow[m];
      const int   iv = irow[m];

      /* normalized |feature diff|, lane = pair */
      const float4 c0123 = *(const float4*)&ctx_s[m * 8];
      const float  c4    = ctx_s[m * 8 + 4];
      float f0 = fabsf(p0 - c0123.x) * inv0;
      float f1 = fabsf(p1 - c0123.y) * inv0;
      float f2 = fabsf(p2 - c0123.z) * inv2;
      float f3 = fabsf(p3 - c0123.w) * inv2;
      float x  = fabsf(p4 - c4) * inv4;
      float fc = x - floorf(x);
      float f4 = fminf(fc, 1.0f - fc);

      const int q0 = pk2(f0, f1), q1 = pk2(f2, f3), q2 = pk2(f4, 0.0f);
      const bool lo = (hf == 0);

      /* A tile 0: pairs 0..15 (already resident in lanes 0..15, K=0..7 slots) */
      V16U A0;
      #pragma unroll
      for (int k = 0; k < 8; ++k) A0.i[k] = 0;
      A0.i[0] = lo ? q0 : 0;  A0.i[1] = lo ? q1 : 0;  A0.i[2] = lo ? q2 : 0;

      /* A tile 1: pairs 16..31, pulled down with one xor-16 shuffle per dword */
      const int s0 = __shfl_xor(q0, 16);
      const int s1 = __shfl_xor(q1, 16);
      const int s2 = __shfl_xor(q2, 16);
      V16U A1;
      #pragma unroll
      for (int k = 0; k < 8; ++k) A1.i[k] = 0;
      A1.i[0] = lo ? s0 : 0;  A1.i[1] = lo ? s1 : 0;  A1.i[2] = lo ? s2 : 0;

      /* fn @ [W1|Wa] + [b1|ba]  -> D[pair][0..3]=pre-relu h, D[pair][4]=score */
      v8f d0 = __builtin_amdgcn_wmma_f32_16x16x32_f16(false, A0.h, false, Bm.h,
                                                      (short)0, cbias, false, false);
      v8f d1 = __builtin_amdgcn_wmma_f32_16x16x32_f16(false, A1.h, false, Bm.h,
                                                      (short)0, cbias, false, false);

      /* transpose D back to pair-per-lane through LDS scratch */
      #pragma unroll
      for (int r = 0; r < 8; ++r) {
        wbase[(hf * 8 + r) * DROW + col]        = d0[r];
        wbase[(16 + hf * 8 + r) * DROW + col]   = d1[r];
      }
      asm volatile("s_wait_dscnt 0" ::: "memory");

      const float4 hv = *(const float4*)&wbase[lane * DROW];
      const float  sc = wbase[lane * DROW + 4];

      const float h0 = fmaxf(hv.x, 0.f), h1 = fmaxf(hv.y, 0.f);
      const float h2 = fmaxf(hv.z, 0.f), h3 = fmaxf(hv.w, 0.f);
      const float u  = h0 * w2v0 + h1 * w2v1 + h2 * w2v2 + h3 * w2v3 + b2v;
      /* fast sigmoid: single v_rcp_f32 instead of IEEE div sequence */
      const float e  = __builtin_amdgcn_rcpf(1.0f + __expf(-u));

      /* branch-free masked softmax accumulation */
      const float ex = __expf(sc);
      const bool valid = (iv != 0) & (dv <= thr);   /* no short-circuit */
      const float w = valid ? ex : 0.0f;            /* cndmask, not exec branch */
      denom += w;
      numer += w * e;
    }

    /* wave32 reduction over the M partials */
    #pragma unroll
    for (int off = 16; off > 0; off >>= 1) {
      denom += __shfl_xor(denom, off);
      numer += __shfl_xor(numer, off);
    }
    if (lane == 0) {
      float r = (denom > 0.0f) ? (numer / denom) : 0.0f;
      if (!isfinite(r)) r = 0.0f;
      out[(size_t)b * Ndim + n] = r;
    }
  }
}

extern "C" void kernel_launch(void* const* d_in, const int* in_sizes, int n_in,
                              void* d_out, int out_size, void* d_ws, size_t ws_size,
                              hipStream_t stream) {
  const float* points = (const float*)d_in[0];
  /* d_in[1] = points_mask   (unused by reference math) */
  const float* ctxp   = (const float*)d_in[2];
  /* d_in[3] = context_points_mask (unused by reference math) */
  const float* dist   = (const float*)d_in[4];
  const unsigned char* inter = (const unsigned char*)d_in[5];
  const int*   gmd    = (const int*)d_in[6];
  const float* W1 = (const float*)d_in[7];
  const float* b1 = (const float*)d_in[8];
  const float* W2 = (const float*)d_in[9];
  const float* b2 = (const float*)d_in[10];
  const float* Wa = (const float*)d_in[11];
  const float* ba = (const float*)d_in[12];

  dim3 grid(Bdim * (Ndim / ROWS_PER_WG));
  similarity_kernel<<<grid, THREADS, 0, stream>>>(points, ctxp, dist, inter, gmd,
                                                  W1, b1, W2, b2, Wa, ba,
                                                  (float*)d_out);
}
